// LampSignatureEncoder_77799037599905
// MI455X (gfx1250) — compile-verified, matrix operand order checked by reading, not run on hardware
//
#include <hip/hip_runtime.h>
#include <hip/hip_bf16.h>

typedef __attribute__((ext_vector_type(16))) _Float16 v16h;
typedef __attribute__((ext_vector_type(8)))  _Float16 v8h;
typedef __attribute__((ext_vector_type(8)))  float    v8f;

#define FIN 128

// ---------------------------------------------------------------------------
// Degree / normalization kernels
// ---------------------------------------------------------------------------
__global__ void gcn_deg_init(float* __restrict__ deg, int N) {
    int i = blockIdx.x * blockDim.x + threadIdx.x;
    if (i < N) deg[i] = 1.0f;  // self-loop contributes 1 to every node's degree
}

__global__ void gcn_deg_count(const int* __restrict__ dst, float* __restrict__ deg, int E) {
    int e = blockIdx.x * blockDim.x + threadIdx.x;
    if (e < E) atomicAdd(&deg[dst[e]], 1.0f);
}

__global__ void gcn_dinv(const float* __restrict__ deg, float* __restrict__ dinv, int N) {
    int i = blockIdx.x * blockDim.x + threadIdx.x;
    if (i < N) dinv[i] = __frsqrt_rn(deg[i]);  // deg >= 1 always (self loops)
}

// ---------------------------------------------------------------------------
// One-time weight prep: W[K,NOUT] f32 (row-major) -> Wt[NOUT,K] f16.
// Makes B fragments contiguous along K (two b128 loads per K-step).
// ---------------------------------------------------------------------------
template <int NOUT>
__global__ void gcn_prep_wt(const float* __restrict__ W, _Float16* __restrict__ Wt) {
    int t = blockIdx.x * blockDim.x + threadIdx.x;  // over FIN*NOUT
    if (t < FIN * NOUT) {
        int k = t / NOUT;
        int n = t - k * NOUT;
        Wt[(size_t)n * FIN + k] = (_Float16)W[t];
    }
}

// ---------------------------------------------------------------------------
// WMMA GEMM:  Hs = (A[M,128] @ W[128,NOUT]) * dinv[row]   (row-scaled)
// Also writes Agg = Hs (self-loop initialization of the accumulator).
// One wave computes one 16x16 output tile; K=128 in 4 steps of 16x16x32 f16.
// B comes from the pre-transposed f16 Wt[NOUT,128].
// ---------------------------------------------------------------------------
template <int NOUT>
__global__ void gcn_gemm_wmma(const float* __restrict__ A,
                              const _Float16* __restrict__ Wt,
                              const float* __restrict__ dinv,
                              float* __restrict__ Hs,
                              float* __restrict__ Agg,
                              int M) {
    const int lane  = threadIdx.x & 31;
    const int wave  = threadIdx.x >> 5;
    const int tilesN   = NOUT / 16;
    const int numTiles = ((M + 15) >> 4) * tilesN;
    const int tile = blockIdx.x * (blockDim.x >> 5) + wave;
    if (tile >= numTiles) return;

    const int tm = tile / tilesN;
    const int tn = tile - tm * tilesN;
    const int m0 = tm << 4;
    const int n0 = tn << 4;

    const int khalf = lane >> 4;        // 0: lanes 0-15, 1: lanes 16-31
    const int rrel  = lane & 15;
    const int row   = m0 + rrel;
    const int rowc  = row < M ? row : (M - 1);   // clamp => EXEC stays all-1 for WMMA
    const int col   = n0 + rrel;

    v8f c = {};
    const float*    arow = A + (size_t)rowc * FIN;
    const _Float16* brow = Wt + (size_t)col * FIN;

#pragma unroll
    for (int kt = 0; kt < FIN / 32; ++kt) {
        const int kb = kt * 32 + khalf * 8;
        v16h a, b;
        const float* pa = arow + kb;
#pragma unroll
        for (int i = 0; i < 8; ++i) {
            a[i]     = (_Float16)pa[i];        // K = kb .. kb+7
            a[i + 8] = (_Float16)pa[16 + i];   // K = kb+16 .. kb+23
        }
        const v8h blo = *(const v8h*)(brow + kb);        // K = kb .. kb+7
        const v8h bhi = *(const v8h*)(brow + kb + 16);   // K = kb+16 .. kb+23
#pragma unroll
        for (int i = 0; i < 8; ++i) {
            b[i]     = blo[i];
            b[i + 8] = bhi[i];
        }
        c = __builtin_amdgcn_wmma_f32_16x16x32_f16(
                /*neg_a=*/false, a, /*neg_b=*/false, b,
                /*c_mod=*/(short)0, c, /*reuse_a=*/false, /*reuse_b=*/false);
    }

    // C/D layout: VGPR j -> row m0 + j + 8*khalf, col = n0 + (lane&15)
#pragma unroll
    for (int j = 0; j < 8; ++j) {
        const int r = m0 + j + khalf * 8;
        if (r < M) {
            const float v = c[j] * dinv[r];
            const size_t o = (size_t)r * NOUT + col;
            Hs[o]  = v;   // scaled features for message passing
            Agg[o] = v;   // accumulator initialized with self-loop term
        }
    }
}

// ---------------------------------------------------------------------------
// Edge scatter: F/32 waves per edge, each wave handles 32 contiguous features.
// The edge index is wave-uniform (forced into SGPRs via readfirstlane), so
// src/dst become scalar loads; the vector pipe does only the coalesced gather
// of Hs[src] and the coalesced atomic-add segment into Agg[dst].
// Hs/Agg (~51MB each) + edge list fit in the 192MB L2 -> L2-resident atomics.
// ---------------------------------------------------------------------------
template <int F>
__global__ void gcn_edge_scatter(const int* __restrict__ src,
                                 const int* __restrict__ dst,
                                 const float* __restrict__ Hs,
                                 float* __restrict__ Agg,
                                 int E) {
    constexpr unsigned WPE = F / 32;  // waves per edge
    const int      lane = threadIdx.x & 31;
    const unsigned wloc = (unsigned)__builtin_amdgcn_readfirstlane(threadIdx.x >> 5);
    const unsigned wid  = blockIdx.x * (blockDim.x >> 5) + wloc;
    const unsigned e    = wid / WPE;
    const unsigned fseg = wid - e * WPE;
    if (e >= (unsigned)E) return;
    const int s = src[e];           // uniform -> s_load
    const int d = dst[e];           // uniform -> s_load
    const int f = (int)(fseg * 32) + lane;
    atomicAdd(&Agg[(size_t)d * F + f], Hs[(size_t)s * F + f]);
}

// ---------------------------------------------------------------------------
// Post-aggregation epilogue: x = dinv[node] * x + bias[f]  (optional relu),
// in place.
// ---------------------------------------------------------------------------
template <int F, int LOGF, bool RELU>
__global__ void gcn_epilogue(float* __restrict__ X,
                             const float* __restrict__ dinv,
                             const float* __restrict__ bias,
                             long long total) {
    long long t = (long long)blockIdx.x * blockDim.x + threadIdx.x;
    if (t >= total) return;
    const int       f = (int)(t & (F - 1));
    const long long i = t >> LOGF;
    float v = dinv[i] * X[t] + bias[f];
    if (RELU) v = v > 0.0f ? v : 0.0f;
    X[t] = v;
}

// ---------------------------------------------------------------------------
// Launch
// ---------------------------------------------------------------------------
extern "C" void kernel_launch(void* const* d_in, const int* in_sizes, int n_in,
                              void* d_out, int out_size, void* d_ws, size_t ws_size,
                              hipStream_t stream) {
    const float* x  = (const float*)d_in[0];
    const int*   ei = (const int*)  d_in[1];   // [2, E] int32
    const float* w1 = (const float*)d_in[2];   // [128, 128]
    const float* b1 = (const float*)d_in[3];   // [128]
    const float* w2 = (const float*)d_in[4];   // [128, 64]
    const float* b2 = (const float*)d_in[5];   // [64]
    float*       out = (float*)d_out;          // [N, 64]

    const int N = in_sizes[0] / FIN;
    const int E = in_sizes[1] / 2;
    const int* src = ei;       // edge_index[0]
    const int* dst = ei + E;   // edge_index[1]

    // Workspace carve-up (~103.3 MB):
    char* ws = (char*)d_ws;
    float* deg  = (float*)ws;  ws += sizeof(float) * (size_t)N;
    float* dinv = (float*)ws;  ws += sizeof(float) * (size_t)N;
    float* bufA = (float*)ws;  ws += sizeof(float) * (size_t)N * 128;   // Hs1 / Ts2
    float* bufB = (float*)ws;  ws += sizeof(float) * (size_t)N * 128;   // agg1 -> H
    _Float16* wt1 = (_Float16*)ws;  ws += sizeof(_Float16) * 128 * 128; // W1^T f16
    _Float16* wt2 = (_Float16*)ws;                                      // W2^T f16

    const int TB = 256;

    // 0) weight prep (tiny, once per call)
    gcn_prep_wt<128><<<(128 * 128 + TB - 1) / TB, TB, 0, stream>>>(w1, wt1);
    gcn_prep_wt<64><<<(128 * 64 + TB - 1) / TB, TB, 0, stream>>>(w2, wt2);

    // 1) degrees + dinv
    gcn_deg_init<<<(N + TB - 1) / TB, TB, 0, stream>>>(deg, N);
    gcn_deg_count<<<(E + TB - 1) / TB, TB, 0, stream>>>(dst, deg, E);
    gcn_dinv<<<(N + TB - 1) / TB, TB, 0, stream>>>(deg, dinv, N);

    // 2) layer 1 GEMM (128 -> 128), epilogue scales by dinv[row], inits agg
    {
        const int tiles  = ((N + 15) >> 4) * (128 / 16);
        const int wavesB = TB / 32;
        gcn_gemm_wmma<128><<<(tiles + wavesB - 1) / wavesB, TB, 0, stream>>>(
            x, wt1, dinv, bufA, bufB, N);
    }

    // 3) layer 1 edge aggregation (atomic scatter-add, L2 resident)
    {
        const long long threads = (long long)E * 128;
        gcn_edge_scatter<128><<<(unsigned)((threads + TB - 1) / TB), TB, 0, stream>>>(
            src, dst, bufA, bufB, E);
    }

    // 4) layer 1 epilogue: H = relu(dinv * agg + b1)  (in place in bufB)
    {
        const long long total = (long long)N * 128;
        gcn_epilogue<128, 7, true><<<(unsigned)((total + TB - 1) / TB), TB, 0, stream>>>(
            bufB, dinv, b1, total);
    }

    // 5) layer 2 GEMM (128 -> 64), epilogue scales by dinv[row], inits d_out
    {
        const int tiles  = ((N + 15) >> 4) * (64 / 16);
        const int wavesB = TB / 32;
        gcn_gemm_wmma<64><<<(tiles + wavesB - 1) / wavesB, TB, 0, stream>>>(
            bufB, wt2, dinv, bufA, out, N);
    }

    // 6) layer 2 edge aggregation into d_out
    {
        const long long threads = (long long)E * 64;
        gcn_edge_scatter<64><<<(unsigned)((threads + TB - 1) / TB), TB, 0, stream>>>(
            src, dst, bufA, out, E);
    }

    // 7) final epilogue: out = dinv * out + b2 (no relu), in place
    {
        const long long total = (long long)N * 64;
        gcn_epilogue<64, 6, false><<<(unsigned)((total + TB - 1) / TB), TB, 0, stream>>>(
            out, dinv, b2, total);
    }
}
